// MutualInformation_66838281060468
// MI455X (gfx1250) — matrix operand order repaired and checked by reading, not compiled, not run on hardware
//
#include <hip/hip_runtime.h>

typedef _Float16 v16h __attribute__((ext_vector_type(16)));
typedef _Float16 v8h  __attribute__((ext_vector_type(8)));
typedef float    v8f  __attribute__((ext_vector_type(8)));

#define BATCH 8
#define LEN   65536
#define NB    256
#define EPSF  1e-10f
#define SIGINV 10.0f      // 1/sigma, sigma = 0.1

// ---- joint GEMM (fused exp + WMMA) ----
#define TILE  128         // output tile per workgroup (128x128)
#define NC    32          // L-slices per (batch, tile)
#define SLICE (LEN / NC)  // 2048
#define KCH   (SLICE / 32)
#define SA    40          // smA stride in halves ([m][k], padded)
#define SB    136         // smB stride in halves ([k][n], padded)

__global__ __launch_bounds__(256) void mi_joint_wmma(
    const float* __restrict__ x, const float* __restrict__ y,
    float* __restrict__ J)
{
  __shared__ _Float16 smA[TILE * SA];   // kx tile: [m_local 0..127][k 0..31]
  __shared__ _Float16 smB[32 * SB];     // ky tile transposed: [k 0..31][n_local 0..127]

  const int idx = blockIdx.x;
  const int b   = idx >> 7;            // 128 blocks per batch
  const int rem = idx & 127;
  const int mt  = rem >> 6;            // 0..1
  const int nt  = (rem >> 5) & 1;      // 0..1
  const int sl  = rem & 31;            // 0..31
  const int m_base = mt * TILE;
  const int n_base = nt * TILE;
  const long base = (long)b * LEN + (long)sl * SLICE;

  const int t    = threadIdx.x;
  const int lane = t & 31;
  const int wave = t >> 5;
  const int wm   = wave & 1;    // 2 waves along M -> 4 m-subtiles each
  const int wn   = wave >> 1;   // 4 waves along N -> 2 n-subtiles each

  v8f acc[4][2] = {};

  const int kpA = t & 15;       // k-pair for smA build
  const int mg  = t >> 4;       // 16 groups x 8 bins
  const int kB  = t & 31;       // k for smB build
  const int ng  = t >> 5;       // 8 groups x 16 bins

  for (int kc = 0; kc < KCH; ++kc) {
    const long l0 = base + (long)kc * 32;
    __syncthreads();            // previous iteration's fragment reads done

    // build kx tile: smA[m][k]
    {
      const float v0 = x[l0 + 2 * kpA]     * 255.0f;
      const float v1 = x[l0 + 2 * kpA + 1] * 255.0f;
      #pragma unroll
      for (int i = 0; i < 8; ++i) {
        const int   ml = mg * 8 + i;
        const float bv = (float)(m_base + ml);
        const float d0 = (v0 - bv) * SIGINV;
        const float d1 = (v1 - bv) * SIGINV;
        union { _Float16 h[2]; unsigned u; } pk;
        pk.h[0] = (_Float16)__expf(-0.5f * d0 * d0);
        pk.h[1] = (_Float16)__expf(-0.5f * d1 * d1);
        *reinterpret_cast<unsigned*>(&smA[ml * SA + 2 * kpA]) = pk.u;
      }
    }
    // build ky tile transposed: smB[k][n]
    {
      const float v = y[l0 + kB] * 255.0f;
      #pragma unroll
      for (int j = 0; j < 8; ++j) {
        const int   nl = ng * 16 + 2 * j;
        const float b0 = (float)(n_base + nl);
        const float d0 = (v - b0) * SIGINV;
        const float d1 = (v - b0 - 1.0f) * SIGINV;
        union { _Float16 h[2]; unsigned u; } pk;
        pk.h[0] = (_Float16)__expf(-0.5f * d0 * d0);
        pk.h[1] = (_Float16)__expf(-0.5f * d1 * d1);
        *reinterpret_cast<unsigned*>(&smB[kB * SB + nl]) = pk.u;
      }
    }
    __syncthreads();

    // A fragments: lane<16 holds K {k0..k0+7, k0+16..k0+23} with k0=0; lanes>=16 k0=8
    const int k0   = (lane & 16) ? 8 : 0;
    const int mrow = lane & 15;
    v16h afr[4];
    #pragma unroll
    for (int i = 0; i < 4; ++i) {
      const _Float16* pa = &smA[((wm * 4 + i) * 16 + mrow) * SA];
      v8h lo = *reinterpret_cast<const v8h*>(pa + k0);
      v8h hi = *reinterpret_cast<const v8h*>(pa + k0 + 16);
      afr[i] = __builtin_shufflevector(lo, hi, 0,1,2,3,4,5,6,7,8,9,10,11,12,13,14,15);
    }
    // B fragments: lane column N = lane&15, K {k0..k0+7, k0+16..k0+23} (same as A)
    v16h bfr[2];
    #pragma unroll
    for (int j = 0; j < 2; ++j) {
      const _Float16* pb = &smB[(wn * 2 + j) * 16 + mrow];
      #pragma unroll
      for (int s = 0; s < 8; ++s) {
        bfr[j][s]     = pb[(k0 + s) * SB];
        bfr[j][s + 8] = pb[(k0 + 16 + s) * SB];
      }
    }
    #pragma unroll
    for (int i = 0; i < 4; ++i)
      #pragma unroll
      for (int j = 0; j < 2; ++j)
        acc[i][j] = __builtin_amdgcn_wmma_f32_16x16x32_f16(
            false, afr[i], false, bfr[j], (short)0, acc[i][j], false, false);
  }

  // accumulate partial 128x128 tile into J[b] (C/D layout: lanes<16 M=r, lanes>=16 M=r+8)
  const int Nl = lane & 15;
  const int Mo = (lane & 16) ? 8 : 0;
  #pragma unroll
  for (int i = 0; i < 4; ++i) {
    #pragma unroll
    for (int j = 0; j < 2; ++j) {
      const int row0 = m_base + (wm * 4 + i) * 16 + Mo;
      const int col  = n_base + (wn * 2 + j) * 16 + Nl;
      #pragma unroll
      for (int r = 0; r < 8; ++r)
        atomicAdd(&J[((long)b << 16) + (long)(row0 + r) * NB + col], acc[i][j][r]);
    }
  }
}

// ---- marginal raw sums: p_raw[which][b][m] = sum_l exp(...) ----
__global__ __launch_bounds__(256) void mi_marginal(
    const float* __restrict__ x, const float* __restrict__ y,
    float* __restrict__ p_raw)
{
  __shared__ float red[256];
  const int idx   = blockIdx.x;
  const int which = idx >> 11;
  const int rem   = idx & 2047;
  const int b     = rem >> 8;
  const int m     = rem & 255;
  const float* src = which ? y : x;
  const int t = threadIdx.x;
  const float bv = (float)m;
  const long base = (long)b * LEN;
  float s = 0.0f;
  for (int i = 0; i < LEN / 256; ++i) {
    const float v = src[base + i * 256 + t] * 255.0f;
    const float d = (v - bv) * SIGINV;
    s += __expf(-0.5f * d * d);
  }
  red[t] = s; __syncthreads();
  for (int st = 128; st > 0; st >>= 1) {
    if (t < st) red[t] += red[t + st];
    __syncthreads();
  }
  if (t == 0) p_raw[which * (BATCH * NB) + b * NB + m] = red[0];
}

// ---- entropies + MI ----
__device__ __forceinline__ float blk_reduce(float v, float* red) {
  const int t = threadIdx.x;
  red[t] = v; __syncthreads();
  for (int st = 128; st > 0; st >>= 1) {
    if (t < st) red[t] += red[t + st];
    __syncthreads();
  }
  const float r = red[0];
  __syncthreads();
  return r;
}

__global__ __launch_bounds__(256) void mi_finalize(
    const float* __restrict__ J, const float* __restrict__ p_raw,
    float* __restrict__ out)
{
  __shared__ float red[256];
  const int b = blockIdx.x;
  const int t = threadIdx.x;

  const float rx = p_raw[b * NB + t] * (1.0f / LEN);          // mean over L
  const float Sx = blk_reduce(rx, red);
  const float px = rx / (Sx + EPSF);
  const float Hx = blk_reduce(-px * __log2f(px + EPSF), red);

  const float ry = p_raw[BATCH * NB + b * NB + t] * (1.0f / LEN);
  const float Sy = blk_reduce(ry, red);
  const float py = ry / (Sy + EPSF);
  const float Hy = blk_reduce(-py * __log2f(py + EPSF), red);

  const float* Jb = J + ((long)b << 16);
  float sj = 0.0f;
  for (int i = 0; i < NB; ++i) sj += Jb[i * NB + t];
  const float Sj  = blk_reduce(sj, red);
  const float inv = 1.0f / (Sj + EPSF);
  float hj = 0.0f;
  for (int i = 0; i < NB; ++i) {
    const float v = Jb[i * NB + t] * inv;
    hj += -v * __log2f(v + EPSF);
  }
  const float Hxy = blk_reduce(hj, red);

  if (t == 0) {
    float mi = Hx + Hy - Hxy;
    mi = 2.0f * mi / (Hx + Hy);
    out[b] = mi;
  }
}

extern "C" void kernel_launch(void* const* d_in, const int* in_sizes, int n_in,
                              void* d_out, int out_size, void* d_ws, size_t ws_size,
                              hipStream_t stream) {
  const float* x = (const float*)d_in[0];
  const float* y = (const float*)d_in[1];
  float* ws    = (float*)d_ws;
  float* J     = ws;                          // 8*256*256 floats
  float* p_raw = ws + BATCH * NB * NB;        // 2*8*256 floats

  const size_t zero_bytes = (size_t)(BATCH * NB * NB + 2 * BATCH * NB) * sizeof(float);
  hipMemsetAsync(d_ws, 0, zero_bytes, stream);

  mi_joint_wmma<<<dim3(BATCH * 2 * 2 * NC), dim3(256), 0, stream>>>(x, y, J);
  mi_marginal <<<dim3(2 * BATCH * NB),      dim3(256), 0, stream>>>(x, y, p_raw);
  mi_finalize <<<dim3(BATCH),               dim3(256), 0, stream>>>(J, p_raw, (float*)d_out);
}
